// cheb_conv_with_SAt_36051955482859
// MI455X (gfx1250) — matrix operand order repaired and checked by reading, not compiled, not run on hardware
//
#include <hip/hip_runtime.h>

#define BATCH 32
#define NN    1024
#define FFD   64
#define TTD   12
#define KORD  3
#define OOD   64
#define FT    768   // F*T columns of stage-1 GEMM
#define NCH   32    // n-chunk == WMMA K depth for bf16

typedef __attribute__((ext_vector_type(8)))  float  v8f;
typedef __attribute__((ext_vector_type(16))) __bf16 v16bf;

union FragB16 {
    unsigned short u[16];
    uint4          q[2];
    v16bf          v;
};

static __device__ __forceinline__ unsigned short f2bf(float f) {
    unsigned int u = __float_as_uint(f);
    unsigned int r = u + 0x7FFFu + ((u >> 16) & 1u);   // round-to-nearest-even
    return (unsigned short)(r >> 16);
}

// ---------------------------------------------------------------------------
// Prep kernel: x (B,N,FT) f32  ->  xT (B,FT,N) bf16, LDS-tiled 32x32 transpose
// so both the global read (along c) and write (along n) are coalesced.
// ---------------------------------------------------------------------------
__global__ __launch_bounds__(256)
void xpose_bf16_kernel(const float* __restrict__ x,
                       unsigned short* __restrict__ xT)
{
    __shared__ float tile[32][33];          // +1 pad kills bank conflicts
    const int tid = threadIdx.x;
    const int c0 = blockIdx.x * 32;
    const int n0 = blockIdx.y * 32;
    const int b  = blockIdx.z;

    const int tr = tid >> 5;                // 0..7
    const int tc = tid & 31;                // 0..31

    #pragma unroll
    for (int i = 0; i < 4; ++i) {
        int n = n0 + tr + i * 8;
        tile[tr + i * 8][tc] = x[((size_t)b * NN + n) * FT + c0 + tc];
    }
    __syncthreads();
    #pragma unroll
    for (int i = 0; i < 4; ++i) {
        int c = c0 + tr + i * 8;
        xT[((size_t)b * FT + c) * NN + n0 + tc] = f2bf(tile[tc][tr + i * 8]);
    }
}

// ---------------------------------------------------------------------------
// Main fused kernel. PREPACKED=true : B-fragments are 2x b128 bf16 loads from
// xT (no conversion VALU).  PREPACKED=false : fallback, converts x on the fly.
// ---------------------------------------------------------------------------
template <bool PREPACKED>
__global__ __launch_bounds__(256)
void cheb_conv_sat_wmma_kernel(const float* __restrict__ x,            // (B,N,F,T) f32
                               const unsigned short* __restrict__ xT,  // (B,FT,N) bf16
                               const float* __restrict__ sat,          // (B,N,N)
                               const float* __restrict__ cheb,         // (K,N,N)
                               const float* __restrict__ theta,        // (K,F,O)
                               float* __restrict__ out)                // (B,N,O,T)
{
    __shared__ unsigned short sPs[KORD][16 * NCH];      // [k][m][n]   3 KB
    __shared__ unsigned short sRs[TTD * 16 * FFD];      // [t][m][f]  24 KB
    __shared__ unsigned short sTh[KORD][OOD * FFD];     // [k][o][f]  24 KB

    const int tid   = threadIdx.x;
    const int wave  = tid >> 5;
    const int lane  = tid & 31;
    const int lhalf = lane >> 4;      // which half-wave
    const int l16   = lane & 15;
    const int kb    = lhalf * 16;     // K-base for A-fragment packing

    const int m0 = blockIdx.x * 16;   // output-row tile
    const int b  = blockIdx.y;        // batch

    // ---- stage Theta^T into LDS as bf16: sTh[k][o*F + f] ----
    #pragma unroll 1
    for (int e = tid; e < KORD * FFD * OOD; e += 256) {
        int k = e / (FFD * OOD);
        int r = e - k * (FFD * OOD);
        int f = r / OOD;
        int o = r - f * OOD;
        sTh[k][o * FFD + f] = f2bf(theta[e]);
    }

    const v8f vzero = {0.f, 0.f, 0.f, 0.f, 0.f, 0.f, 0.f, 0.f};
    v8f acc1[KORD][6];
    #pragma unroll
    for (int k = 0; k < KORD; ++k)
        #pragma unroll
        for (int j = 0; j < 6; ++j)
            acc1[k][j] = vzero;

    const float*          xb  = x  + (size_t)b * NN * FT;
    const unsigned short* xtb = xT + (size_t)b * FT * NN;

    // ===================== Stage 1: rhs_k = (T_k*A_b)^T @ X =====================
    #pragma unroll 1
    for (int nc = 0; nc < NN / NCH; ++nc) {
        const int n0 = nc * NCH;

        // cooperative: P[k][m][n] = bf16(cheb[k,n,m0+m] * sat[b,n,m0+m])
        #pragma unroll
        for (int e = tid; e < 16 * NCH; e += 256) {
            int nl = e >> 4;           // local n
            int mm = e & 15;           // local m
            float a = sat[((size_t)b * NN + n0 + nl) * NN + m0 + mm];
            #pragma unroll
            for (int k = 0; k < KORD; ++k) {
                float c = cheb[((size_t)k * NN + n0 + nl) * NN + m0 + mm];
                sPs[k][mm * NCH + nl] = f2bf(a * c);
            }
        }
        __syncthreads();

        // A-fragments (shared across waves, per k): 2x ds_read_b128 each
        FragB16 af[KORD];
        #pragma unroll
        for (int k = 0; k < KORD; ++k) {
            const uint4* pA = (const uint4*)&sPs[k][l16 * NCH + kb];
            af[k].q[0] = pA[0];   // K = kb .. kb+7
            af[k].q[1] = pA[1];   // K = kb+8 .. kb+15
        }

        // each wave: 6 column tiles of 16 (f,t)-columns
        #pragma unroll
        for (int j = 0; j < 6; ++j) {
            const int cb = (wave * 6 + j) * 16;
            FragB16 bfr;
            if constexpr (PREPACKED) {
                // 16 contiguous bf16 = rows n0+lhalf*16 .. +15 at column cb+l16
                const uint4* pxt = (const uint4*)
                    &xtb[((size_t)(cb + l16)) * NN + n0 + lhalf * 16];
                bfr.q[0] = pxt[0];
                bfr.q[1] = pxt[1];
                // prefetch next n-chunk's fragment (+32 bf16 = +64B)
                __builtin_prefetch((const char*)pxt + 64, 0, 3);
            } else {
                const float* px =
                    xb + (size_t)(n0 + lhalf * 16) * FT + cb + l16;
                #pragma unroll
                for (int i = 0; i < 8; ++i) {       // rows n = 2i, 2i+1 of half
                    float lo = px[(size_t)(2 * i)     * FT];
                    float hi = px[(size_t)(2 * i + 1) * FT];
                    bfr.u[2 * i]     = f2bf(lo);
                    bfr.u[2 * i + 1] = f2bf(hi);
                }
            }
            #pragma unroll
            for (int k = 0; k < KORD; ++k)
                acc1[k][j] = __builtin_amdgcn_wmma_f32_16x16x32_bf16(
                    false, af[k].v, false, bfr.v, (short)0, acc1[k][j],
                    false, false);
        }
        __syncthreads();   // protect sPs before next chunk overwrites it
    }

    // ===================== Stage 2: out += rhs_k @ Theta_k =====================
    v8f oacc[6];
    #pragma unroll
    for (int j = 0; j < 6; ++j) oacc[j] = vzero;

    #pragma unroll 1
    for (int k = 0; k < KORD; ++k) {
        // spill rhs_k tile to LDS as bf16 in [t][m][f] layout
        #pragma unroll
        for (int j = 0; j < 6; ++j) {
            int c = (wave * 6 + j) * 16 + l16;   // (f,t) column this lane holds
            int f = c / TTD;
            int t = c - f * TTD;
            #pragma unroll
            for (int r = 0; r < 8; ++r) {
                int m = r + lhalf * 8;           // C-layout row
                sRs[(t * 16 + m) * FFD + f] = f2bf(acc1[k][j][r]);
            }
        }
        __syncthreads();

        // each wave: 6 output tiles (t, o-block); contract over f = 2 halves of 32
        #pragma unroll
        for (int j = 0; j < 6; ++j) {
            int tt = wave * 6 + j;
            int t  = tt >> 2;
            int ob = (tt & 3) * 16;
            #pragma unroll
            for (int h = 0; h < 2; ++h) {
                FragB16 aF, bF;
                const uint4* pA =
                    (const uint4*)&sRs[(t * 16 + l16) * FFD + h * 32 + kb];
                aF.q[0] = pA[0];   // f = h*32 + kb .. +7
                aF.q[1] = pA[1];   // f = h*32 + kb + 8 .. +15
                const uint4* pB =
                    (const uint4*)&sTh[k][(ob + l16) * FFD + h * 32 + lhalf * 16];
                bF.q[0] = pB[0];   // f rows 0..7 of half
                bF.q[1] = pB[1];   // f rows 8..15 of half
                oacc[j] = __builtin_amdgcn_wmma_f32_16x16x32_bf16(
                    false, aF.v, false, bF.v, (short)0, oacc[j], false, false);
            }
        }
        __syncthreads();   // protect sRs before next k overwrites it
    }

    // ===================== ReLU + store (B,N,O,T) f32 =====================
    #pragma unroll
    for (int j = 0; j < 6; ++j) {
        int tt = wave * 6 + j;
        int t  = tt >> 2;
        int o  = (tt & 3) * 16 + l16;
        #pragma unroll
        for (int r = 0; r < 8; ++r) {
            int m = r + lhalf * 8;
            float v = oacc[j][r];
            v = v > 0.f ? v : 0.f;
            out[(((size_t)b * NN + m0 + m) * OOD + o) * TTD + t] = v;
        }
    }
}

extern "C" void kernel_launch(void* const* d_in, const int* in_sizes, int n_in,
                              void* d_out, int out_size, void* d_ws, size_t ws_size,
                              hipStream_t stream) {
    const float* x    = (const float*)d_in[0];   // (B,N,F,T)
    const float* sat  = (const float*)d_in[1];   // (B,N,N)
    const float* cheb = (const float*)d_in[2];   // (K,N,N)
    const float* th   = (const float*)d_in[3];   // (K,F,O)
    float* out = (float*)d_out;                  // (B,N,O,T)

    const size_t need = (size_t)BATCH * FT * NN * sizeof(unsigned short); // 48 MB
    dim3 grid(NN / 16, BATCH);

    if (ws_size >= need) {
        unsigned short* xT = (unsigned short*)d_ws;
        dim3 g1(FT / 32, NN / 32, BATCH);
        xpose_bf16_kernel<<<g1, 256, 0, stream>>>(x, xT);
        cheb_conv_sat_wmma_kernel<true><<<grid, 256, 0, stream>>>(
            x, xT, sat, cheb, th, out);
    } else {
        cheb_conv_sat_wmma_kernel<false><<<grid, 256, 0, stream>>>(
            x, (const unsigned short*)nullptr, sat, cheb, th, out);
    }
}
